// ProteinMPNNModule_33535104647901
// MI455X (gfx1250) — matrix operand order repaired and checked by reading, not compiled, not run on hardware
//
#include <hip/hip_runtime.h>
#include <math.h>

// ---------------- constants ----------------
constexpr int B_ = 2;
constexpr int L_ = 1024;
constexpr int H_ = 128;
constexpr int K_ = 48;
constexpr int NEDGE = B_ * L_ * K_;   // 98304
constexpr int NNODE = B_ * L_;        // 2048
constexpr int NRBF = 16;
constexpr int NPOS = 16;
constexpr int EIN = NPOS + 25 * NRBF; // 416
constexpr int MAXREL = 32;
constexpr int VOCAB = 21;

typedef _Float16 f16;
typedef _Float16 v16h __attribute__((ext_vector_type(16)));
typedef float v8f __attribute__((ext_vector_type(8)));

__constant__ int PAIR_A[24] = {0,2,3,4,1,1,1,1,0,0,0,4,4,3,0,2,3,4,2,3,4,2,3,2};
__constant__ int PAIR_B[24] = {0,2,3,4,0,2,3,4,2,3,4,2,3,2,1,1,1,1,0,0,0,4,4,3};

__device__ __forceinline__ float gelu_f(float x) {
  return 0.5f * x * (1.0f + erff(x * 0.70710678118654752f));
}

__device__ __forceinline__ float block_sum128(float v, float* red, int t) {
  red[t] = v;
  __syncthreads();
#pragma unroll
  for (int s = 64; s > 0; s >>= 1) {
    if (t < s) red[t] += red[t + s];
    __syncthreads();
  }
  float r = red[0];
  __syncthreads();
  return r;
}

// ---------------- tiny utility kernels ----------------
__global__ void mpnn_cast_k(const float* __restrict__ src, f16* __restrict__ dst, int n) {
  int i = blockIdx.x * blockDim.x + threadIdx.x;
  if (i < n) dst[i] = (f16)src[i];
}

// cast + transpose: src is [Kd][N] f32 (row-major din x dout), dst is [N][Kd] f16
__global__ void mpnn_castT_k(const float* __restrict__ src, f16* __restrict__ dst,
                             int Kd, int N) {
  int idx = blockIdx.x * blockDim.x + threadIdx.x;
  if (idx >= Kd * N) return;
  int k = idx / N;
  int n = idx % N;
  dst[(size_t)n * Kd + k] = (f16)src[idx];
}

__global__ void mpnn_zero_k(float* __restrict__ p, int n) {
  int i = blockIdx.x * blockDim.x + threadIdx.x;
  if (i < n) p[i] = 0.0f;
}

// ---------------- geometry ----------------
__global__ void mpnn_backbone_k(const float* __restrict__ coords, float* __restrict__ atoms) {
  int idx = blockIdx.x * blockDim.x + threadIdx.x;
  if (idx >= NNODE) return;
  const float* c = coords + (size_t)idx * 12;
  float N[3], Ca[3], Cc[3], O[3], bv[3], cv[3], a[3];
#pragma unroll
  for (int d = 0; d < 3; ++d) { N[d] = c[d]; Ca[d] = c[3 + d]; Cc[d] = c[6 + d]; O[d] = c[9 + d]; }
#pragma unroll
  for (int d = 0; d < 3; ++d) { bv[d] = Ca[d] - N[d]; cv[d] = Cc[d] - Ca[d]; }
  a[0] = bv[1] * cv[2] - bv[2] * cv[1];
  a[1] = bv[2] * cv[0] - bv[0] * cv[2];
  a[2] = bv[0] * cv[1] - bv[1] * cv[0];
  float* o = atoms + (size_t)idx * 15;
#pragma unroll
  for (int d = 0; d < 3; ++d) {
    o[0 + d] = N[d];
    o[3 + d] = Ca[d];
    o[6 + d] = Cc[d];
    o[9 + d] = O[d];
    o[12 + d] = -0.58273431f * a[d] + 0.56802827f * bv[d] - 0.54067466f * cv[d] + Ca[d];
  }
}

// one block per node: full row of adjusted distances + iterative top-K (smallest, tie->lowest idx)
__global__ __launch_bounds__(256) void mpnn_topk_k(const float* __restrict__ atoms,
                                                   const float* __restrict__ mask,
                                                   int* __restrict__ E_idx,
                                                   float* __restrict__ Dnb) {
  __shared__ float dist[L_];
  __shared__ float rv[256];
  __shared__ int ri[256];
  int node = blockIdx.x;
  int b = node / L_;
  int t = threadIdx.x;
  float mi = mask[node];
  const float* cai = atoms + (size_t)node * 15 + 3;
  float cx = cai[0], cy = cai[1], cz = cai[2];
  for (int j = t; j < L_; j += 256) {
    const float* caj = atoms + (size_t)(b * L_ + j) * 15 + 3;
    float dx = cx - caj[0], dy = cy - caj[1], dz = cz - caj[2];
    float m2 = mi * mask[b * L_ + j];
    dist[j] = m2 * sqrtf(dx * dx + dy * dy + dz * dz + 1e-6f);
  }
  __syncthreads();
  float lm = -1e30f;
  for (int j = t; j < L_; j += 256) lm = fmaxf(lm, dist[j]);
  rv[t] = lm;
  __syncthreads();
  for (int s = 128; s > 0; s >>= 1) {
    if (t < s) rv[t] = fmaxf(rv[t], rv[t + s]);
    __syncthreads();
  }
  float dmax = rv[0];
  __syncthreads();
  for (int j = t; j < L_; j += 256) {
    float m2 = mi * mask[b * L_ + j];
    dist[j] = dist[j] + (1.0f - m2) * dmax;
  }
  __syncthreads();
  for (int s = 0; s < K_; ++s) {
    float bvv = 3.4e38f;
    int bii = L_;
    for (int j = t; j < L_; j += 256) {
      float v = dist[j];
      if (v < bvv || (v == bvv && j < bii)) { bvv = v; bii = j; }
    }
    rv[t] = bvv; ri[t] = bii;
    __syncthreads();
    for (int sz = 128; sz > 0; sz >>= 1) {
      if (t < sz) {
        float v2 = rv[t + sz]; int i2 = ri[t + sz];
        if (v2 < rv[t] || (v2 == rv[t] && i2 < ri[t])) { rv[t] = v2; ri[t] = i2; }
      }
      __syncthreads();
    }
    if (t == 0) {
      int e = node * K_ + s;
      E_idx[e] = ri[0];
      Dnb[e] = rv[0];
      dist[ri[0]] = 3.4e38f;
    }
    __syncthreads();
  }
}

// one block per edge: 25 distances -> 416 f16 features; also mask_attend
__global__ __launch_bounds__(256) void mpnn_edgefeat_k(
    const float* __restrict__ atoms, const int* __restrict__ E_idx,
    const float* __restrict__ Dnb, const float* __restrict__ mask,
    const int* __restrict__ residx, const int* __restrict__ chain,
    const float* __restrict__ pos_w, const float* __restrict__ pos_b,
    f16* __restrict__ Ebuf, float* __restrict__ mask_att) {
  __shared__ float d25[25];
  int e = blockIdx.x;
  int b = e / (L_ * K_);
  int i = (e / K_) % L_;
  int j = E_idx[e];
  int t = threadIdx.x;
  if (t == 0) {
    d25[0] = Dnb[e];
    mask_att[e] = mask[b * L_ + i] * mask[b * L_ + j];
  }
  if (t >= 1 && t < 25) {
    int p = t - 1;
    const float* pi = atoms + (size_t)(b * L_ + i) * 15 + PAIR_A[p] * 3;
    const float* pj = atoms + (size_t)(b * L_ + j) * 15 + PAIR_B[p] * 3;
    float dx = pi[0] - pj[0], dy = pi[1] - pj[1], dz = pi[2] - pj[2];
    d25[t] = sqrtf(dx * dx + dy * dy + dz * dz + 1e-6f);
  }
  __syncthreads();
  int off = residx[b * L_ + i] - residx[b * L_ + j];
  int sc = (chain[b * L_ + i] == chain[b * L_ + j]) ? 1 : 0;
  int dclip = off + MAXREL;
  if (dclip < 0) dclip = 0;
  if (dclip > 2 * MAXREL) dclip = 2 * MAXREL;
  int dd = sc ? dclip : (2 * MAXREL + 1);
  f16* row = Ebuf + (size_t)e * EIN;
  for (int f = t; f < EIN; f += 256) {
    float v;
    if (f < NPOS) {
      v = pos_w[dd * NPOS + f] + pos_b[f];
    } else {
      int p = (f - NPOS) >> 4;
      int r = (f - NPOS) & 15;
      float mu = 2.0f + (20.0f / 15.0f) * (float)r;
      float z = (d25[p] - mu) / 1.25f;
      v = expf(-z * z);
    }
    row[f] = (f16)v;
  }
}

// ---------------- WMMA GEMM: C[M,N] = A[M,Kd] * W[Kd,N] (+bias)(+gelu) ----------------
// Weights are pre-transposed: Wt[n*Kd + k].  Double-buffered LDS, 1 barrier / K-step.
// All 8 B fragments are preloaded so the 8 WMMAs issue back-to-back.
union FragU { v16h v; uint4 q[2]; };

template <bool GELU, bool OUTF16>
__global__ __launch_bounds__(256) void mpnn_gemm_k(const f16* __restrict__ A,
                                                   const f16* __restrict__ Wt,
                                                   const float* __restrict__ bias,
                                                   void* __restrict__ Cout,
                                                   int M, int Kd, int N) {
  __shared__ __align__(16) f16 As[2][128 * 32];
  __shared__ __align__(16) f16 Bs[2][128 * 32];  // Bs[n*32 + k]
  const int tid = threadIdx.x;
  const int lane = tid & 31;
  const int wave = tid >> 5;
  const int g = lane >> 4;
  const int l16 = lane & 15;
  const int rowBlk = blockIdx.x * 128;
  const int colBlk = blockIdx.y * 128;
  (void)M;

  v8f acc[8];
#pragma unroll
  for (int j = 0; j < 8; ++j) {
#pragma unroll
    for (int r = 0; r < 8; ++r) acc[j][r] = 0.0f;
  }

  const int r128 = tid >> 1;        // A row / Wt column-row (0..127)
  const int c16 = (tid & 1) << 4;   // 0 or 16 (k offset within 32-wide tile)

  const f16* aptr = A + (size_t)(rowBlk + r128) * Kd + c16;
  const f16* bptr = Wt + (size_t)(colBlk + r128) * Kd + c16;

  // prologue: tile 0 -> buffer 0
  uint4 a0 = ((const uint4*)aptr)[0];
  uint4 a1 = ((const uint4*)aptr)[1];
  uint4 b0 = ((const uint4*)bptr)[0];
  uint4 b1 = ((const uint4*)bptr)[1];
  {
    uint4* da = (uint4*)(As[0] + r128 * 32 + c16);
    da[0] = a0; da[1] = a1;
    uint4* db = (uint4*)(Bs[0] + r128 * 32 + c16);
    db[0] = b0; db[1] = b1;
  }
  __syncthreads();

  const int nIter = Kd >> 5;
  for (int it = 0; it < nIter; ++it) {
    const int cur = it & 1;
    if (it + 1 < nIter) {  // issue next-tile global loads before the WMMA chain
      const f16* ap = aptr + (size_t)(it + 1) * 32;
      const f16* bp = bptr + (size_t)(it + 1) * 32;
      a0 = ((const uint4*)ap)[0];
      a1 = ((const uint4*)ap)[1];
      b0 = ((const uint4*)bp)[0];
      b1 = ((const uint4*)bp)[1];
      if (it + 2 < nIter) {  // gfx1250 global_prefetch two tiles ahead
        __builtin_prefetch(aptr + (size_t)(it + 2) * 32, 0, 0);
        __builtin_prefetch(bptr + (size_t)(it + 2) * 32, 0, 0);
      }
    }
    // preload A fragment + all 8 B fragments, then fire the WMMA chain back-to-back
    FragU af;
    {
      const f16* asrc = As[cur] + (wave * 16 + l16) * 32;
      af.q[0] = ((const uint4*)(asrc + g * 16))[0];
      af.q[1] = ((const uint4*)(asrc + g * 16 + 8))[0];
    }
    FragU bf[8];
#pragma unroll
    for (int j = 0; j < 8; ++j) {
      const uint4* bp2 = (const uint4*)(Bs[cur] + (j * 16 + l16) * 32 + g * 16);
      bf[j].q[0] = bp2[0];
      bf[j].q[1] = bp2[1];
    }
#pragma unroll
    for (int j = 0; j < 8; ++j) {
      acc[j] = __builtin_amdgcn_wmma_f32_16x16x32_f16(false, af.v, false, bf[j].v,
                                                      (short)0, acc[j], false, false);
    }
    if (it + 1 < nIter) {
      uint4* da = (uint4*)(As[cur ^ 1] + r128 * 32 + c16);
      da[0] = a0; da[1] = a1;
      uint4* db = (uint4*)(Bs[cur ^ 1] + r128 * 32 + c16);
      db[0] = b0; db[1] = b1;
      __syncthreads();  // buffer written here was last read at iteration it-1
    }
  }

#pragma unroll
  for (int j = 0; j < 8; ++j) {
    int n = colBlk + j * 16 + l16;
    float bv = bias ? bias[n] : 0.0f;
#pragma unroll
    for (int r = 0; r < 8; ++r) {
      int m = rowBlk + wave * 16 + g * 8 + r;
      float v = acc[j][r] + bv;
      if (GELU) v = gelu_f(v);
      if (OUTF16)
        ((f16*)Cout)[(size_t)m * N + n] = (f16)v;
      else
        ((float*)Cout)[(size_t)m * N + n] = v;
    }
  }
}

// ---------------- layernorm / residual kernels (width 128, one row per block) ----------------
__global__ __launch_bounds__(128) void mpnn_ln_rows_f16_k(const float* __restrict__ X,
                                                          const float* __restrict__ s,
                                                          const float* __restrict__ b,
                                                          f16* __restrict__ Y) {
  __shared__ float red[128];
  int row = blockIdx.x, t = threadIdx.x;
  float x = X[(size_t)row * H_ + t];
  float mu = block_sum128(x, red, t) * (1.0f / H_);
  float d = x - mu;
  float var = block_sum128(d * d, red, t) * (1.0f / H_);
  Y[(size_t)row * H_ + t] = (f16)(d * rsqrtf(var + 1e-5f) * s[t] + b[t]);
}

__global__ __launch_bounds__(128) void mpnn_res_ln_k(const float* __restrict__ X,
                                                     const float* __restrict__ R,
                                                     const float* __restrict__ s,
                                                     const float* __restrict__ b,
                                                     const float* __restrict__ rowmask,
                                                     float* __restrict__ Y) {
  __shared__ float red[128];
  int row = blockIdx.x, t = threadIdx.x;
  float x = X[(size_t)row * H_ + t] + R[(size_t)row * H_ + t];
  float mu = block_sum128(x, red, t) * (1.0f / H_);
  float d = x - mu;
  float var = block_sum128(d * d, red, t) * (1.0f / H_);
  float y = d * rsqrtf(var + 1e-5f) * s[t] + b[t];
  if (rowmask) y *= rowmask[row];
  Y[(size_t)row * H_ + t] = y;
}

__global__ __launch_bounds__(128) void mpnn_node_msg_ln_k(float* __restrict__ hV,
                                                          const float* __restrict__ m,
                                                          const float* __restrict__ mask_att,
                                                          const float* __restrict__ s,
                                                          const float* __restrict__ b,
                                                          int useAtt) {
  __shared__ float red[128];
  int node = blockIdx.x, t = threadIdx.x;
  float acc = 0.0f;
  for (int k = 0; k < K_; ++k) {
    size_t e = (size_t)node * K_ + k;
    float w = useAtt ? mask_att[e] : 1.0f;
    acc += w * m[e * H_ + t];
  }
  float x = hV[(size_t)node * H_ + t] + acc * (1.0f / 30.0f);
  float mu = block_sum128(x, red, t) * (1.0f / H_);
  float d = x - mu;
  float var = block_sum128(d * d, red, t) * (1.0f / H_);
  hV[(size_t)node * H_ + t] = d * rsqrtf(var + 1e-5f) * s[t] + b[t];
}

// ---------------- pack (concat) kernels, write f16 GEMM inputs ----------------
__global__ void mpnn_pack_enc_k(const float* __restrict__ hV, const float* __restrict__ hE,
                                const int* __restrict__ E_idx, f16* __restrict__ out) {
  size_t idx = (size_t)blockIdx.x * blockDim.x + threadIdx.x;
  size_t total = (size_t)NEDGE * 384;
  if (idx >= total) return;
  int c = (int)(idx % 384);
  size_t e = idx / 384;
  int node = (int)(e / K_);
  float v;
  if (c < H_) {
    v = hV[(size_t)node * H_ + c];
  } else if (c < 2 * H_) {
    v = hE[e * H_ + (c - H_)];
  } else {
    int b = node / L_;
    int j = E_idx[e];
    v = hV[(size_t)(b * L_ + j) * H_ + (c - 2 * H_)];
  }
  out[e * 384 + c] = (f16)v;
}

__global__ void mpnn_pack_dec_k(const float* __restrict__ hV, const float* __restrict__ hVenc,
                                const float* __restrict__ hE, const int* __restrict__ E_idx,
                                const int* __restrict__ aatype, const float* __restrict__ Ws,
                                const float* __restrict__ mask, const int* __restrict__ inv,
                                f16* __restrict__ out) {
  size_t idx = (size_t)blockIdx.x * blockDim.x + threadIdx.x;
  size_t total = (size_t)NEDGE * 512;
  if (idx >= total) return;
  int c = (int)(idx % 512);
  size_t e = idx / 512;
  int node = (int)(e / K_);
  int b = node / L_;
  int j = E_idx[e];
  int nj = b * L_ + j;
  float v;
  if (c < H_) {
    v = hV[(size_t)node * H_ + c];
  } else {
    float mi = mask[node];
    if (c < 2 * H_) {
      v = mi * hE[e * H_ + (c - H_)];
    } else {
      float att = (inv[node] > inv[nj]) ? 1.0f : 0.0f;
      if (c < 3 * H_)
        v = mi * att * Ws[aatype[nj] * H_ + (c - 2 * H_)];
      else
        v = mi * att * hV[(size_t)nj * H_ + (c - 3 * H_)] +
            mi * (1.0f - att) * hVenc[(size_t)nj * H_ + (c - 3 * H_)];
    }
  }
  out[e * 512 + c] = (f16)v;
}

// ---------------- decoding order ranks ----------------
__global__ void mpnn_invrank_k(const float* __restrict__ mask, const float* __restrict__ randn,
                               int* __restrict__ inv) {
  int idx = blockIdx.x * blockDim.x + threadIdx.x;
  if (idx >= NNODE) return;
  int b = idx / L_;
  int i = idx % L_;
  float ki = (mask[idx] + 1e-4f) * fabsf(randn[idx]);
  int r = 0;
  for (int j = 0; j < L_; ++j) {
    float kj = (mask[b * L_ + j] + 1e-4f) * fabsf(randn[b * L_ + j]);
    if (kj < ki || (kj == ki && j < i)) r++;
  }
  inv[idx] = r;
}

// ---------------- output head: logits, log_softmax, masked NLL accumulation ----------------
__global__ __launch_bounds__(32) void mpnn_nll_k(const float* __restrict__ hV,
                                                 const float* __restrict__ wout,
                                                 const float* __restrict__ bout,
                                                 const int* __restrict__ aatype,
                                                 const float* __restrict__ mask,
                                                 float* __restrict__ accum) {
  __shared__ float lg[VOCAB];
  int node = blockIdx.x, t = threadIdx.x;
  if (t < VOCAB) {
    float s = bout[t];
    const float* h = hV + (size_t)node * H_;
    for (int c = 0; c < H_; ++c) s += h[c] * wout[c * VOCAB + t];
    lg[t] = s;
  }
  __syncthreads();
  if (t == 0) {
    float mx = -1e30f;
    for (int k = 0; k < VOCAB; ++k) mx = fmaxf(mx, lg[k]);
    float se = 0.0f;
    for (int k = 0; k < VOCAB; ++k) se += expf(lg[k] - mx);
    float lse = mx + logf(se);
    float nll = lse - lg[aatype[node]];
    atomicAdd(&accum[0], nll * mask[node]);
    atomicAdd(&accum[1], mask[node]);
  }
}

__global__ void mpnn_final_k(const float* __restrict__ accum, float* __restrict__ out) {
  if (blockIdx.x == 0 && threadIdx.x == 0) out[0] = accum[0] / (accum[1] + 1e-6f);
}

// ---------------- host launch ----------------
extern "C" void kernel_launch(void* const* d_in, const int* in_sizes, int n_in,
                              void* d_out, int out_size, void* d_ws, size_t ws_size,
                              hipStream_t stream) {
  (void)in_sizes; (void)n_in; (void)out_size; (void)ws_size;
  const float* coords = (const float*)d_in[0];
  const int* aatype = (const int*)d_in[1];
  const float* mask = (const float*)d_in[2];
  const int* residx = (const int*)d_in[3];
  const int* chain = (const int*)d_in[4];
  const float* randn = (const float*)d_in[5];
  auto Pf = [&](int i) { return (const float*)d_in[i]; };

  // JAX pytree (sorted dict keys) leaf indices:
  const int I_We_b = 6, I_We_w = 7, I_Wout_b = 8, I_Wout_w = 9, I_Ws = 10;
  auto DB = [](int l) { return 11 + 14 * l; };   // dec layer leaf base
  const int I_eemb_w = 53;
  auto EB = [](int l) { return 54 + 22 * l; };   // enc layer leaf base
  const int I_ne_b = 120, I_ne_s = 121, I_pos_b = 122, I_pos_w = 123;

  char* base = (char*)d_ws;
  size_t off = 0;
  auto alloc = [&](size_t bytes) -> void* {
    void* p = base + off;
    off += (bytes + 255) & ~(size_t)255;
    return p;
  };
  // cast + transpose a [Kd][N] f32 weight into [N][Kd] f16
  auto castwT = [&](int idx, int Kd, int N) -> f16* {
    int n = Kd * N;
    f16* p = (f16*)alloc((size_t)n * sizeof(f16));
    mpnn_castT_k<<<(n + 255) / 256, 256, 0, stream>>>(Pf(idx), p, Kd, N);
    return p;
  };

  // weights -> f16, transposed for the GEMM B-tile path
  f16* w_eemb = castwT(I_eemb_w, EIN, H_);
  f16* w_We = castwT(I_We_w, H_, H_);
  f16 *eW1[3], *eW2[3], *eW3[3], *eW11[3], *eW12[3], *eW13[3], *eFin[3], *eFout[3];
  for (int l = 0; l < 3; ++l) {
    int b0 = EB(l);
    eFin[l] = castwT(b0 + 1, H_, 4 * H_);
    eFout[l] = castwT(b0 + 3, 4 * H_, H_);
    eW1[l] = castwT(b0 + 5, 3 * H_, H_);
    eW11[l] = castwT(b0 + 7, 3 * H_, H_);
    eW12[l] = castwT(b0 + 9, H_, H_);
    eW13[l] = castwT(b0 + 11, H_, H_);
    eW2[l] = castwT(b0 + 13, H_, H_);
    eW3[l] = castwT(b0 + 15, H_, H_);
  }
  f16 *dW1[3], *dW2[3], *dW3[3], *dFin[3], *dFout[3];
  for (int l = 0; l < 3; ++l) {
    int b0 = DB(l);
    dFin[l] = castwT(b0 + 1, H_, 4 * H_);
    dFout[l] = castwT(b0 + 3, 4 * H_, H_);
    dW1[l] = castwT(b0 + 5, 4 * H_, H_);
    dW2[l] = castwT(b0 + 7, H_, H_);
    dW3[l] = castwT(b0 + 9, H_, H_);
  }

  // activations / scratch
  float* atoms = (float*)alloc((size_t)NNODE * 15 * sizeof(float));
  int* E_idx = (int*)alloc((size_t)NEDGE * sizeof(int));
  float* Dnb = (float*)alloc((size_t)NEDGE * sizeof(float));
  float* mask_att = (float*)alloc((size_t)NEDGE * sizeof(float));
  int* invr = (int*)alloc((size_t)NNODE * sizeof(int));
  f16* buf_in = (f16*)alloc((size_t)NEDGE * 512 * sizeof(f16));    // E(416)/h_EV(384)/dec(512)
  f16* buf_mid = (f16*)alloc((size_t)NEDGE * H_ * sizeof(f16));
  f16* buf_mid2 = (f16*)alloc((size_t)NEDGE * H_ * sizeof(f16));
  float* edge_f32 = (float*)alloc((size_t)NEDGE * H_ * sizeof(float));
  float* hE = (float*)alloc((size_t)NEDGE * H_ * sizeof(float));
  float* hV = (float*)alloc((size_t)NNODE * H_ * sizeof(float));
  float* hVenc = (float*)alloc((size_t)NNODE * H_ * sizeof(float));
  f16* node_in = (f16*)alloc((size_t)NNODE * H_ * sizeof(f16));
  f16* node_mid = (f16*)alloc((size_t)NNODE * 4 * H_ * sizeof(f16));
  float* node_out = (float*)alloc((size_t)NNODE * H_ * sizeof(float));
  float* accum = (float*)alloc(2 * sizeof(float));

  auto gemm = [&](const f16* A, const f16* Wt, const float* bias, void* C,
                  int M, int Kd, int N, bool gl, bool outf16) {
    dim3 g(M / 128, N / 128), blk(256);
    if (gl && outf16)
      mpnn_gemm_k<true, true><<<g, blk, 0, stream>>>(A, Wt, bias, C, M, Kd, N);
    else if (!gl && outf16)
      mpnn_gemm_k<false, true><<<g, blk, 0, stream>>>(A, Wt, bias, C, M, Kd, N);
    else if (gl && !outf16)
      mpnn_gemm_k<true, false><<<g, blk, 0, stream>>>(A, Wt, bias, C, M, Kd, N);
    else
      mpnn_gemm_k<false, false><<<g, blk, 0, stream>>>(A, Wt, bias, C, M, Kd, N);
  };

  // ---- features ----
  mpnn_backbone_k<<<(NNODE + 255) / 256, 256, 0, stream>>>(coords, atoms);
  mpnn_topk_k<<<NNODE, 256, 0, stream>>>(atoms, mask, E_idx, Dnb);
  mpnn_edgefeat_k<<<NEDGE, 256, 0, stream>>>(atoms, E_idx, Dnb, mask, residx, chain,
                                             Pf(I_pos_w), Pf(I_pos_b), buf_in, mask_att);
  gemm(buf_in, w_eemb, nullptr, edge_f32, NEDGE, EIN, H_, false, false);
  mpnn_ln_rows_f16_k<<<NEDGE, 128, 0, stream>>>(edge_f32, Pf(I_ne_s), Pf(I_ne_b), buf_mid);
  gemm(buf_mid, w_We, Pf(I_We_b), hE, NEDGE, H_, H_, false, false);
  mpnn_zero_k<<<(NNODE * H_ + 255) / 256, 256, 0, stream>>>(hV, NNODE * H_);

  const int PACK3 = (int)(((size_t)NEDGE * 384 + 255) / 256);
  const int PACK4 = (int)(((size_t)NEDGE * 512 + 255) / 256);

  // ---- encoder ----
  for (int l = 0; l < 3; ++l) {
    int b0 = EB(l);
    mpnn_pack_enc_k<<<PACK3, 256, 0, stream>>>(hV, hE, E_idx, buf_in);
    gemm(buf_in, eW1[l], Pf(b0 + 4), buf_mid, NEDGE, 3 * H_, H_, true, true);
    gemm(buf_mid, eW2[l], Pf(b0 + 12), buf_mid2, NEDGE, H_, H_, true, true);
    gemm(buf_mid2, eW3[l], Pf(b0 + 14), edge_f32, NEDGE, H_, H_, false, false);
    mpnn_node_msg_ln_k<<<NNODE, 128, 0, stream>>>(hV, edge_f32, mask_att,
                                                  Pf(b0 + 17), Pf(b0 + 16), 1);
    mpnn_cast_k<<<(NNODE * H_ + 255) / 256, 256, 0, stream>>>(hV, node_in, NNODE * H_);
    gemm(node_in, eFin[l], Pf(b0 + 0), node_mid, NNODE, H_, 4 * H_, true, true);
    gemm(node_mid, eFout[l], Pf(b0 + 2), node_out, NNODE, 4 * H_, H_, false, false);
    mpnn_res_ln_k<<<NNODE, 128, 0, stream>>>(hV, node_out, Pf(b0 + 19), Pf(b0 + 18), mask, hV);
    mpnn_pack_enc_k<<<PACK3, 256, 0, stream>>>(hV, hE, E_idx, buf_in);
    gemm(buf_in, eW11[l], Pf(b0 + 6), buf_mid, NEDGE, 3 * H_, H_, true, true);
    gemm(buf_mid, eW12[l], Pf(b0 + 8), buf_mid2, NEDGE, H_, H_, true, true);
    gemm(buf_mid2, eW13[l], Pf(b0 + 10), edge_f32, NEDGE, H_, H_, false, false);
    mpnn_res_ln_k<<<NEDGE, 128, 0, stream>>>(hE, edge_f32, Pf(b0 + 21), Pf(b0 + 20), nullptr, hE);
  }

  // ---- decoder ----
  hipMemcpyAsync(hVenc, hV, (size_t)NNODE * H_ * sizeof(float),
                 hipMemcpyDeviceToDevice, stream);
  mpnn_invrank_k<<<(NNODE + 255) / 256, 256, 0, stream>>>(mask, randn, invr);
  for (int l = 0; l < 3; ++l) {
    int b0 = DB(l);
    mpnn_pack_dec_k<<<PACK4, 256, 0, stream>>>(hV, hVenc, hE, E_idx, aatype, Pf(I_Ws),
                                               mask, invr, buf_in);
    gemm(buf_in, dW1[l], Pf(b0 + 4), buf_mid, NEDGE, 4 * H_, H_, true, true);
    gemm(buf_mid, dW2[l], Pf(b0 + 6), buf_mid2, NEDGE, H_, H_, true, true);
    gemm(buf_mid2, dW3[l], Pf(b0 + 8), edge_f32, NEDGE, H_, H_, false, false);
    mpnn_node_msg_ln_k<<<NNODE, 128, 0, stream>>>(hV, edge_f32, mask_att,
                                                  Pf(b0 + 11), Pf(b0 + 10), 0);
    mpnn_cast_k<<<(NNODE * H_ + 255) / 256, 256, 0, stream>>>(hV, node_in, NNODE * H_);
    gemm(node_in, dFin[l], Pf(b0 + 0), node_mid, NNODE, H_, 4 * H_, true, true);
    gemm(node_mid, dFout[l], Pf(b0 + 2), node_out, NNODE, 4 * H_, H_, false, false);
    mpnn_res_ln_k<<<NNODE, 128, 0, stream>>>(hV, node_out, Pf(b0 + 13), Pf(b0 + 12), mask, hV);
  }

  // ---- loss ----
  mpnn_zero_k<<<1, 32, 0, stream>>>(accum, 2);
  mpnn_nll_k<<<NNODE, 32, 0, stream>>>(hV, Pf(I_Wout_w), Pf(I_Wout_b), aatype, mask, accum);
  mpnn_final_k<<<1, 1, 0, stream>>>(accum, (float*)d_out);
}
